// TernaryMultiheadAttention_75720273428528
// MI455X (gfx1250) — matrix operand order repaired and checked by reading, not compiled, not run on hardware
//
#include <hip/hip_runtime.h>
#include <hip/hip_bf16.h>

// ---------------- problem constants ----------------
#define D_MODEL   1024
#define NUM_HEADS 16
#define HEAD_DIM  64
#define BATCH     2
#define SEQ       2048
#define MROWS     (BATCH * SEQ)        // 4096
#define WELEM     (D_MODEL * D_MODEL)  // 1048576
#define XELEM     (MROWS * D_MODEL)    // 4194304

typedef __attribute__((ext_vector_type(16))) _Float16 v16h;
typedef __attribute__((ext_vector_type(8)))  _Float16 v8h;
typedef __attribute__((ext_vector_type(8)))  float    v8f;

__device__ __forceinline__ v8f v8f_zero() {
    v8f z = {0.f, 0.f, 0.f, 0.f, 0.f, 0.f, 0.f, 0.f};
    return z;
}

// D = A(16x32 f16) * B(32x16 f16) + C(16x16 f32)
__device__ __forceinline__ v8f wmma_f16(v16h a, v16h b, v8f c) {
    return __builtin_amdgcn_wmma_f32_16x16x32_f16(
        /*neg_a=*/false, a, /*neg_b=*/false, b,
        /*c_mod=*/(short)0, c, /*reuse_a=*/false, /*reuse_b=*/false);
}

// A-fragment (16x32, f16) from row-major matrix, tile base p, leading dim ld (elems).
// Layout used here (not the ISA table order): lane l holds row m=l&15; lane group g=l>>4:
//   halves 0..15 -> K = 16g + 0..15 (contiguous, matches load_b16)
__device__ __forceinline__ v16h load_a16(const _Float16* p, int ld, int lane) {
    const int m = lane & 15, g = lane >> 4;
    const _Float16* r = p + (size_t)m * ld + 16 * g;
    union { v16h w; v8h h[2]; } u;
    u.h[0] = *(const v8h*)(r);
    u.h[1] = *(const v8h*)(r + 8);
    return u.w;
}

// B-fragment (32x16, f16) where B is stored COLUMN-major (i.e. p points at a
// row-major [N,K] matrix = B^T).  lane l holds col n=l&15, K = 16g + 0..15
// -> one contiguous 32-byte read per lane.
__device__ __forceinline__ v16h load_b16(const _Float16* p, int ld, int lane) {
    const int n = lane & 15, g = lane >> 4;
    return *(const v16h*)(p + (size_t)n * ld + 16 * g);
}

// ---------------- stage 1: abs-mean thresholds (deterministic 2-stage) ------
__global__ void abs_partial_kernel(const float* __restrict__ w0, const float* __restrict__ w1,
                                   const float* __restrict__ w2, const float* __restrict__ w3,
                                   float* __restrict__ partials) {
    const int w = blockIdx.y;
    const float* W = (w == 0) ? w0 : (w == 1) ? w1 : (w == 2) ? w2 : w3;
    __shared__ float red[256];
    float s = 0.f;
    for (int i = blockIdx.x * 256 + threadIdx.x; i < WELEM; i += 256 * 256)
        s += fabsf(W[i]);
    red[threadIdx.x] = s;
    __syncthreads();
    for (int st = 128; st > 0; st >>= 1) {
        if ((int)threadIdx.x < st) red[threadIdx.x] += red[threadIdx.x + st];
        __syncthreads();
    }
    if (threadIdx.x == 0) partials[w * 256 + blockIdx.x] = red[0];
}

__global__ void finish_thresh_kernel(const float* __restrict__ partials,
                                     float* __restrict__ thr) {
    const int w = blockIdx.x;
    __shared__ float red[256];
    red[threadIdx.x] = partials[w * 256 + threadIdx.x];
    __syncthreads();
    for (int st = 128; st > 0; st >>= 1) {
        if ((int)threadIdx.x < st) red[threadIdx.x] += red[threadIdx.x + st];
        __syncthreads();
    }
    if (threadIdx.x == 0) thr[w] = red[0] * (1.0f / (float)WELEM);
}

// ---------------- stage 2: ternarize weights -> f16 --------------------------
__global__ void ternarize_kernel(const float* __restrict__ W, const float* __restrict__ thr,
                                 int widx, _Float16* __restrict__ out) {
    const int i = blockIdx.x * 256 + threadIdx.x;
    if (i >= WELEM) return;
    const float t = thr[widx];
    const float v = W[i];
    const float o = (fabsf(v) > t) ? ((v > 0.f) ? 1.f : -1.f) : 0.f;
    out[i] = (_Float16)o;
}

// ---------------- stage 3: f32 -> f16 activation convert ---------------------
__global__ void f32_to_f16_kernel(const float* __restrict__ in, _Float16* __restrict__ out, int n) {
    const int i = blockIdx.x * 256 + threadIdx.x;
    if (i < n) out[i] = (_Float16)in[i];
}

// ---------------- stage 4: WMMA GEMM  C[M,N] = X[M,K] * Wt^T -----------------
// One wave computes a 32x64 output tile: A frags reused 4x, B frags 2x.
// X: f16 row-major [MROWS, D_MODEL].  Wt: ternary f16 row-major [N,K] (== B col-major).
// OUT_MODE 0: f16 row-major [MROWS, D_MODEL]
// OUT_MODE 1: f32 row-major [MROWS, D_MODEL]  (final output)
// OUT_MODE 2: f16 transposed-V layout [B, H, HEAD_DIM, SEQ]
template <int OUT_MODE>
__global__ void __launch_bounds__(128)
gemm_tern_kernel(const _Float16* __restrict__ X, const _Float16* __restrict__ Wt,
                 void* __restrict__ Out, float scale) {
    const int lane = threadIdx.x & 31;
    const int wv   = threadIdx.x >> 5;
    const int id   = blockIdx.x * 4 + wv;       // 2048 waves = (4096/32)*(1024/64)
    const int tn   = id & 15;                   // 16 tiles along N
    const int tm   = id >> 4;                   // 128 tiles along M
    const int m0   = tm * 32, n0 = tn * 64;
    const int g = lane >> 4, nn = lane & 15;

    v8f acc[2][4];
#pragma unroll
    for (int i = 0; i < 2; ++i)
#pragma unroll
        for (int j = 0; j < 4; ++j) acc[i][j] = v8f_zero();

    for (int k0 = 0; k0 < D_MODEL; k0 += 32) {
        const v16h xa0 = load_a16(X + (size_t)m0 * D_MODEL + k0,        D_MODEL, lane);
        const v16h xa1 = load_a16(X + (size_t)(m0 + 16) * D_MODEL + k0, D_MODEL, lane);
        if (k0 + 32 < D_MODEL) {  // global_prefetch_b8 for next k-tile
            __builtin_prefetch(X  + (size_t)(m0 + lane) * D_MODEL + k0 + 32, 0, 3);
            __builtin_prefetch(Wt + (size_t)(n0 + lane) * D_MODEL + k0 + 32, 0, 3);
            __builtin_prefetch(Wt + (size_t)(n0 + 32 + lane) * D_MODEL + k0 + 32, 0, 3);
        }
#pragma unroll
        for (int j = 0; j < 4; ++j) {
            const v16h wb = load_b16(Wt + (size_t)(n0 + 16 * j) * D_MODEL + k0, D_MODEL, lane);
            acc[0][j] = wmma_f16(xa0, wb, acc[0][j]);
            acc[1][j] = wmma_f16(xa1, wb, acc[1][j]);
        }
    }

#pragma unroll
    for (int i = 0; i < 2; ++i)
#pragma unroll
        for (int j = 0; j < 4; ++j)
#pragma unroll
            for (int r = 0; r < 8; ++r) {
                const int m = m0 + 16 * i + r + 8 * g;   // D layout: m = r + 8g
                const int n = n0 + 16 * j + nn;          //           n = lane&15
                const float val = acc[i][j][r] * scale;
                if (OUT_MODE == 0) {
                    ((_Float16*)Out)[(size_t)m * D_MODEL + n] = (_Float16)val;
                } else if (OUT_MODE == 1) {
                    ((float*)Out)[(size_t)m * D_MODEL + n] = val;
                } else {
                    const int bb = m >> 11, s = m & (SEQ - 1);
                    const int hh = n >> 6,  dd = n & (HEAD_DIM - 1);
                    ((_Float16*)Out)[(((size_t)bb * NUM_HEADS + hh) * HEAD_DIM + dd) * SEQ + s] =
                        (_Float16)val;
                }
            }
}

// ---------------- stage 5: flash attention (one wave = 16 q-rows of one b,h) -
// 64-key chunks: 16 WMMAs per chunk against 64 ds_bpermute + 40 exp2 -> half
// the softmax overhead per key vs a 32-key chunk.
__global__ void __launch_bounds__(128)
attn_kernel(const _Float16* __restrict__ Qp, const _Float16* __restrict__ Kp,
            const _Float16* __restrict__ Vt, _Float16* __restrict__ Aout) {
    const int lane = threadIdx.x & 31;
    const int wv   = threadIdx.x >> 5;
    const int id   = blockIdx.x * 4 + wv;            // 0..4095 = B*H*(SEQ/16)
    const int q16  = id & (SEQ / 16 - 1);
    const int h    = (id >> 7) & (NUM_HEADS - 1);
    const int b    = id >> 11;
    const int q0   = q16 * 16;
    const int g = lane >> 4, nn = lane & 15;
    const float L2E = 1.44269504088896340736f;

    __shared__ __align__(64) _Float16 Pl[4][16][64];
    _Float16(*P)[64] = Pl[wv];

    // Q tile resident in registers (scores already scaled: Q was scaled by 0.125)
    const _Float16* Qb = Qp + (size_t)(b * SEQ + q0) * D_MODEL + h * HEAD_DIM;
    const v16h aq0 = load_a16(Qb,      D_MODEL, lane);   // d = 0..31
    const v16h aq1 = load_a16(Qb + 32, D_MODEL, lane);   // d = 32..63

    v8f o[4];
#pragma unroll
    for (int j = 0; j < 4; ++j) o[j] = v8f_zero();
    float rmax[8], rsum[8];
#pragma unroll
    for (int r = 0; r < 8; ++r) { rmax[r] = -3.0e38f; rsum[r] = 0.f; }

    const _Float16* Kb = Kp + (size_t)(b * SEQ) * D_MODEL + h * HEAD_DIM;
    const _Float16* Vb = Vt + ((size_t)b * NUM_HEADS + h) * HEAD_DIM * (size_t)SEQ;

    for (int key0 = 0; key0 < SEQ; key0 += 64) {
        // ---- scores S[16q x 64k] = Q * K^T (K rows contiguous -> perfect B frags)
        const _Float16* Kt = Kb + (size_t)key0 * D_MODEL;
        v8f s[4];
#pragma unroll
        for (int j = 0; j < 4; ++j) {
            const _Float16* Kj = Kt + (size_t)(16 * j) * D_MODEL;
            s[j] = wmma_f16(aq0, load_b16(Kj,      D_MODEL, lane), v8f_zero());
            s[j] = wmma_f16(aq1, load_b16(Kj + 32, D_MODEL, lane), s[j]);
        }

        // ---- online softmax (row = r+8g lives in one 16-lane half; reduce over lanes)
#pragma unroll
        for (int r = 0; r < 8; ++r) {
            float cm = fmaxf(fmaxf(s[0][r], s[1][r]), fmaxf(s[2][r], s[3][r]));
            cm = fmaxf(cm, __shfl_xor(cm, 1, 32));
            cm = fmaxf(cm, __shfl_xor(cm, 2, 32));
            cm = fmaxf(cm, __shfl_xor(cm, 4, 32));
            cm = fmaxf(cm, __shfl_xor(cm, 8, 32));
            const float nm = fmaxf(rmax[r], cm);
            const float f  = exp2f((rmax[r] - nm) * L2E);
            rmax[r] = nm;
            float p[4], ps = 0.f;
#pragma unroll
            for (int j = 0; j < 4; ++j) {
                p[j] = exp2f((s[j][r] - nm) * L2E);
                ps += p[j];
            }
            ps += __shfl_xor(ps, 1, 32);
            ps += __shfl_xor(ps, 2, 32);
            ps += __shfl_xor(ps, 4, 32);
            ps += __shfl_xor(ps, 8, 32);
            rsum[r] = rsum[r] * f + ps;
#pragma unroll
            for (int j = 0; j < 4; ++j) {
                o[j][r] = o[j][r] * f;
                // D-layout -> LDS (row-major 16x64 f16) for A-fragment relayout
                P[r + 8 * g][nn + 16 * j] = (_Float16)p[j];
            }
        }
        // same-wave DS ops are in-order; compiler inserts s_wait_dscnt
        const v16h pa0 = load_a16(&P[0][0],      64, lane);  // keys 0..31 of chunk
        const v16h pa1 = load_a16(&P[0][0] + 32, 64, lane);  // keys 32..63 of chunk

        // ---- O += P[16x64] * V[64k x 64d]; Vt rows contiguous over keys
        const _Float16* Vc = Vb + key0;
#pragma unroll
        for (int j = 0; j < 4; ++j) {
            const _Float16* Vj = Vc + (size_t)(16 * j) * SEQ;
            o[j] = wmma_f16(pa0, load_b16(Vj,      SEQ, lane), o[j]);
            o[j] = wmma_f16(pa1, load_b16(Vj + 32, SEQ, lane), o[j]);
        }
    }

    _Float16* Ob = Aout + (size_t)(b * SEQ + q0) * D_MODEL + h * HEAD_DIM;
#pragma unroll
    for (int r = 0; r < 8; ++r) {
        const float inv = 1.0f / rsum[r];
        const int m = r + 8 * g;
#pragma unroll
        for (int j = 0; j < 4; ++j)
            Ob[(size_t)m * D_MODEL + 16 * j + nn] = (_Float16)(o[j][r] * inv);
    }
}

// ---------------- orchestration ----------------------------------------------
extern "C" void kernel_launch(void* const* d_in, const int* in_sizes, int n_in,
                              void* d_out, int out_size, void* d_ws, size_t ws_size,
                              hipStream_t stream) {
    const float* q  = (const float*)d_in[0];
    const float* k  = (const float*)d_in[1];
    const float* v  = (const float*)d_in[2];
    const float* wq = (const float*)d_in[3];
    const float* wk = (const float*)d_in[4];
    const float* wv = (const float*)d_in[5];
    const float* wo = (const float*)d_in[6];

    char* ws = (char*)d_ws;
    float* partials = (float*)ws;                 // 4*256 floats
    float* thr      = (float*)(ws + 4096);        // 4 floats
    size_t off = 8192;
    _Float16* W16 = (_Float16*)(ws + off); off += (size_t)4 * WELEM * 2;  // 8 MB
    _Float16* X16 = (_Float16*)(ws + off); off += (size_t)XELEM * 2;      // 8 MB (reused)
    _Float16* Qp  = (_Float16*)(ws + off); off += (size_t)XELEM * 2;      // 8 MB
    _Float16* Kp  = (_Float16*)(ws + off); off += (size_t)XELEM * 2;      // 8 MB
    _Float16* Vt  = (_Float16*)(ws + off); off += (size_t)XELEM * 2;      // 8 MB
    _Float16* A16 = X16;  // X16 dead after V projection; reuse for attention output

    // thresholds (deterministic two-stage reduction)
    abs_partial_kernel<<<dim3(256, 4), 256, 0, stream>>>(wq, wk, wv, wo, partials);
    finish_thresh_kernel<<<4, 256, 0, stream>>>(partials, thr);

    // ternarize all four weights into f16 (row-major [N,K] == column-major B)
    ternarize_kernel<<<WELEM / 256, 256, 0, stream>>>(wq, thr, 0, W16 + (size_t)0 * WELEM);
    ternarize_kernel<<<WELEM / 256, 256, 0, stream>>>(wk, thr, 1, W16 + (size_t)1 * WELEM);
    ternarize_kernel<<<WELEM / 256, 256, 0, stream>>>(wv, thr, 2, W16 + (size_t)2 * WELEM);
    ternarize_kernel<<<WELEM / 256, 256, 0, stream>>>(wo, thr, 3, W16 + (size_t)3 * WELEM);

    // Q projection (folds 1/sqrt(HEAD_DIM) = 0.125, exact in f16)
    f32_to_f16_kernel<<<XELEM / 256, 256, 0, stream>>>(q, X16, XELEM);
    gemm_tern_kernel<0><<<512, 128, 0, stream>>>(X16, W16 + (size_t)0 * WELEM, Qp, 0.125f);
    // K projection
    f32_to_f16_kernel<<<XELEM / 256, 256, 0, stream>>>(k, X16, XELEM);
    gemm_tern_kernel<0><<<512, 128, 0, stream>>>(X16, W16 + (size_t)1 * WELEM, Kp, 1.0f);
    // V projection, stored pre-transposed [B,H,Dh,S] for contiguous PV B-fragments
    f32_to_f16_kernel<<<XELEM / 256, 256, 0, stream>>>(v, X16, XELEM);
    gemm_tern_kernel<2><<<512, 128, 0, stream>>>(X16, W16 + (size_t)2 * WELEM, Vt, 1.0f);

    // flash attention: B*H*(SEQ/16) = 4096 waves
    attn_kernel<<<1024, 128, 0, stream>>>(Qp, Kp, Vt, A16);

    // output projection -> f32 d_out
    gemm_tern_kernel<1><<<512, 128, 0, stream>>>(A16, W16 + (size_t)3 * WELEM, d_out, 1.0f);
}
